// DynamicNetworkEvolutionModel_82806969466954
// MI455X (gfx1250) — matrix operand order repaired and checked by reading, not compiled, not run on hardware
//
#include <hip/hip_runtime.h>
#include <hip/hip_bf16.h>

#define NN 22
#define HID 32
#define DM 128

typedef __attribute__((ext_vector_type(16))) _Float16 v16h;
typedef __attribute__((ext_vector_type(8)))  float    v8f;

__device__ __forceinline__ float leaky02(float x) { return x > 0.f ? x : 0.2f * x; }
__device__ __forceinline__ float gelu_exact(float x) {
    return 0.5f * x * (1.f + erff(x * 0.70710678118654752440f));
}

// ---- WMMA fragment helpers (CDNA5 ISA 7.12.2 layouts, wave32) ----
// A: 16x32 f16. lane L holds row = L%16; VGPR v pair: K = (v/4)*16 + (L>=16)*8 + (v%4)*2
__device__ __forceinline__ v16h load_a_frag(const float* src, int ld, int lane) {
    v16h a;
    int row = lane & 15;
    int koff = (lane >> 4) * 8;
#pragma unroll
    for (int v = 0; v < 8; ++v) {
        int k = (v >> 2) * 16 + koff + (v & 3) * 2;
        a[2 * v]     = (_Float16)src[row * ld + k];
        a[2 * v + 1] = (_Float16)src[row * ld + k + 1];
    }
    return a;
}
// B: 32x16 f16. lane L holds col = L%16; VGPR v pair: K = (v/4)*16 + (L>=16)*8 + (v%4)*2
__device__ __forceinline__ v16h load_b_frag(const float* src, int ld, int lane) {
    v16h b;
    int col = lane & 15;
    int koff = (lane >> 4) * 8;
#pragma unroll
    for (int v = 0; v < 8; ++v) {
        int k = (v >> 2) * 16 + koff + (v & 3) * 2;
        b[2 * v]     = (_Float16)src[k * ld + col];
        b[2 * v + 1] = (_Float16)src[(k + 1) * ld + col];
    }
    return b;
}
// D: 16x16 f32. element (m,n): lane = n + 16*(m/8), vgpr = m%8

// =====================  GAT branch  =====================
__global__ __launch_bounds__(256) void gat_kernel(
    const float* __restrict__ bn,
    const float* __restrict__ W1, const float* __restrict__ a1,
    const float* __restrict__ W2, const float* __restrict__ a2,
    float* __restrict__ out32, int f)
{
    __shared__ float s_adj[NN * NN];
    __shared__ float s_feat[NN * HID];
    __shared__ float s_Wh[NN * HID];
    __shared__ float s_src[NN * 4];
    __shared__ float s_dst[NN * 4];
    __shared__ float s_alpha[NN * NN * 4];
    __shared__ float s_out[NN * HID];

    const int m = blockIdx.x;
    const int tid = threadIdx.x;
    const float* base = bn + (size_t)m * NN * NN * 4;

    for (int u = tid; u < NN * NN; u += 256) {
        int i = u / NN, j = u % NN;
        float v = base[(i * NN + j) * 4 + f];
        s_adj[u] = v;
        s_feat[i * HID + j] = v;   // layer-1 node features = adjacency rows (din=22)
    }
    __syncthreads();

    for (int layer = 0; layer < 2; ++layer) {
        const int din = (layer == 0) ? NN : HID;
        const float* W  = (layer == 0) ? W1 : W2;
        const float* av = (layer == 0) ? a1 : a2;

        for (int u = tid; u < NN * HID; u += 256) {
            int n = u / HID, o = u % HID;
            float acc = 0.f;
            for (int k = 0; k < din; ++k) acc += s_feat[n * HID + k] * W[k * HID + o];
            s_Wh[u] = acc;
        }
        __syncthreads();

        for (int u = tid; u < NN * 4; u += 256) {
            int n = u / 4, hh = u % 4;
            float ss = 0.f, dd = 0.f;
            for (int d = 0; d < 8; ++d) {
                float w = s_Wh[n * HID + hh * 8 + d];
                ss += w * av[hh * 16 + d];
                dd += w * av[hh * 16 + 8 + d];
            }
            s_src[u] = ss; s_dst[u] = dd;
        }
        __syncthreads();

        // alpha[i][j][hh] = softmax_i( mask(leaky(src_i + dst_j)) )
        for (int u = tid; u < NN * 4; u += 256) {
            int j = u / 4, hh = u % 4;
            float ev[NN];
            float mx = -3.4e38f;
            for (int i = 0; i < NN; ++i) {
                float e = leaky02(s_src[i * 4 + hh] + s_dst[j * 4 + hh]);
                e = (s_adj[i * NN + j] > 1e-8f) ? e : -1e9f;
                ev[i] = e;
                mx = fmaxf(mx, e);
            }
            float sum = 0.f;
            for (int i = 0; i < NN; ++i) { ev[i] = __expf(ev[i] - mx); sum += ev[i]; }
            float inv = 1.f / sum;
            for (int i = 0; i < NN; ++i) s_alpha[(i * NN + j) * 4 + hh] = ev[i] * inv;
        }
        __syncthreads();

        for (int u = tid; u < NN * HID; u += 256) {
            int j = u / HID, o = u % HID, hh = o / 8;
            float acc = 0.f;
            for (int i = 0; i < NN; ++i)
                acc += s_alpha[(i * NN + j) * 4 + hh] * s_Wh[i * HID + o];
            s_out[u] = fmaxf(acc, 0.f);   // relu
        }
        __syncthreads();
        for (int u = tid; u < NN * HID; u += 256) s_feat[u] = s_out[u];
        __syncthreads();
    }

    for (int o = tid; o < HID; o += 256) {
        float acc = 0.f;
        for (int n = 0; n < NN; ++n) acc += s_feat[n * HID + o];
        out32[((size_t)m * 4 + f) * HID + o] = acc / (float)NN;
    }
}

// =====================  GCN branch  =====================
__global__ __launch_bounds__(256) void gcn_kernel(
    const float* __restrict__ bn,
    const float* __restrict__ W1, const float* __restrict__ b1,
    const float* __restrict__ W2, const float* __restrict__ b2,
    float* __restrict__ out32)
{
    __shared__ float s_adj[NN * NN];
    __shared__ float s_na[NN * NN];
    __shared__ float s_t[NN * HID];
    __shared__ float s_h[NN * HID];
    __shared__ float s_dis[NN];

    const int m = blockIdx.x;
    const int tid = threadIdx.x;
    const float* base = bn + (size_t)m * NN * NN * 4;

    for (int u = tid; u < NN * NN; u += 256) s_adj[u] = base[u * 4 + 3];
    __syncthreads();
    for (int i = tid; i < NN; i += 256) {
        float s = 0.f;
        for (int j = 0; j < NN; ++j) s += s_adj[i * NN + j];
        s_dis[i] = rsqrtf(fmaxf(s, 1e-6f));
    }
    __syncthreads();
    for (int u = tid; u < NN * NN; u += 256) {
        int i = u / NN, j = u % NN;
        s_na[u] = s_adj[u] * s_dis[i] * s_dis[j];
    }
    __syncthreads();
    // t = na @ feat  (feat = adjacency rows, 22 cols)
    for (int u = tid; u < NN * NN; u += 256) {
        int i = u / NN, k = u % NN;
        float acc = 0.f;
        for (int j = 0; j < NN; ++j) acc += s_na[i * NN + j] * s_adj[j * NN + k];
        s_t[i * HID + k] = acc;
    }
    __syncthreads();
    for (int u = tid; u < NN * HID; u += 256) {
        int i = u / HID, o = u % HID;
        float acc = b1[o];
        for (int k = 0; k < NN; ++k) acc += s_t[i * HID + k] * W1[k * HID + o];
        s_h[u] = fmaxf(acc, 0.f);
    }
    __syncthreads();
    for (int u = tid; u < NN * HID; u += 256) {
        int i = u / HID, o = u % HID;
        float acc = 0.f;
        for (int j = 0; j < NN; ++j) acc += s_na[i * NN + j] * s_h[j * HID + o];
        s_t[u] = acc;
    }
    __syncthreads();
    for (int u = tid; u < NN * HID; u += 256) {
        int i = u / HID, o = u % HID;
        float acc = b2[o];
        for (int k = 0; k < HID; ++k) acc += s_t[i * HID + k] * W2[k * HID + o];
        s_h[u] = fmaxf(acc, 0.f);
    }
    __syncthreads();
    for (int o = tid; o < HID; o += 256) {
        float acc = 0.f;
        for (int n = 0; n < NN; ++n) acc += s_h[n * HID + o];
        out32[((size_t)m * 4 + 3) * HID + o] = acc / (float)NN;
    }
}

// =====================  Fusion head (WMMA)  =====================
// One block = 4 graphs = 16 token rows; 4 waves, each owns 2 of 8 column tiles.
__device__ __forceinline__ void gemm_16x128(
    const float* s_A, const float* __restrict__ Wg, const float* __restrict__ bias,
    const float* s_res, float* s_D, int lane, int wvid)
{
#pragma unroll
    for (int i = 0; i < 2; ++i) {
        int ct = wvid * 2 + i;
        v8f c = {};
#pragma unroll
        for (int kk = 0; kk < 4; ++kk) {
            v16h a = load_a_frag(s_A + kk * 32, DM, lane);
            v16h b = load_b_frag(Wg + (kk * 32) * DM + ct * 16, DM, lane);
            c = __builtin_amdgcn_wmma_f32_16x16x32_f16(false, a, false, b, (short)0, c,
                                                       false, false);
        }
        int n = lane & 15, mb = (lane >> 4) * 8;
#pragma unroll
        for (int v = 0; v < 8; ++v) {
            float r = c[v] + bias[ct * 16 + n];
            if (s_res) r += s_res[(mb + v) * DM + ct * 16 + n];
            s_D[(mb + v) * DM + ct * 16 + n] = r;
        }
    }
}

__global__ __launch_bounds__(128) void fuse_kernel(
    const float* __restrict__ out32,
    const float* __restrict__ proj_w, const float* __restrict__ proj_b,
    const float* __restrict__ wq, const float* __restrict__ bq,
    const float* __restrict__ wk, const float* __restrict__ bk,
    const float* __restrict__ wv, const float* __restrict__ bv,
    const float* __restrict__ wo, const float* __restrict__ bo,
    const float* __restrict__ ln1_g, const float* __restrict__ ln1_b,
    const float* __restrict__ gw1, const float* __restrict__ gb1,
    const float* __restrict__ gw2, const float* __restrict__ gb2,
    const float* __restrict__ ln2_g, const float* __restrict__ ln2_b,
    const float* __restrict__ tw1, const float* __restrict__ tb1,
    const float* __restrict__ tw2, const float* __restrict__ tb2,
    const float* __restrict__ srt, float* __restrict__ out)
{
    __shared__ float s_in[16 * HID];
    __shared__ float s_tok[16 * DM];
    __shared__ float s_q[16 * DM];     // later reused for ao2 (post-residual)
    __shared__ float s_k[16 * DM];     // later reused for LN1 output
    __shared__ float s_v[16 * DM];     // later reused for gate hidden [16][64]
    __shared__ float s_ao[16 * DM];
    __shared__ float s_gate[16];
    __shared__ float s_wts[16];
    __shared__ float s_fused[4 * DM];
    __shared__ float s_fln[4 * DM];
    __shared__ float s_hid[4 * 64];

    const int tid = threadIdx.x;
    const int lane = tid & 31;
    const int wvid = tid >> 5;
    const int mbase = blockIdx.x * 4;

    for (int u = tid; u < 16 * HID; u += 128)
        s_in[u] = out32[(size_t)mbase * 4 * HID + u];
    __syncthreads();

    // tokens = out32(16x32) @ proj_w(32x128) + proj_b  -- one WMMA per tile (K=32)
    {
        v16h a = load_a_frag(s_in, HID, lane);
#pragma unroll
        for (int i = 0; i < 2; ++i) {
            int ct = wvid * 2 + i;
            v16h b = load_b_frag(proj_w + ct * 16, DM, lane);
            v8f c = {};
            c = __builtin_amdgcn_wmma_f32_16x16x32_f16(false, a, false, b, (short)0, c,
                                                       false, false);
            int n = lane & 15, mb = (lane >> 4) * 8;
#pragma unroll
            for (int v = 0; v < 8; ++v)
                s_tok[(mb + v) * DM + ct * 16 + n] = c[v] + proj_b[ct * 16 + n];
        }
    }
    __syncthreads();

    gemm_16x128(s_tok, wq, bq, nullptr, s_q, lane, wvid);
    gemm_16x128(s_tok, wk, bk, nullptr, s_k, lane, wvid);
    gemm_16x128(s_tok, wv, bv, nullptr, s_v, lane, wvid);
    __syncthreads();

    // 4-token, 4-head attention (per graph) on VALU
    if (tid < 16) {
        int g = tid >> 2, hh = tid & 3;
        const float scale = 0.17677669529663687f;  // 1/sqrt(32)
        float at[4][4];
        for (int tq = 0; tq < 4; ++tq) {
            float mx = -3.4e38f;
            for (int tk = 0; tk < 4; ++tk) {
                float acc = 0.f;
                for (int d = 0; d < 32; ++d)
                    acc += s_q[(g * 4 + tq) * DM + hh * 32 + d] *
                           s_k[(g * 4 + tk) * DM + hh * 32 + d];
                at[tq][tk] = acc * scale;
                mx = fmaxf(mx, at[tq][tk]);
            }
            float sum = 0.f;
            for (int tk = 0; tk < 4; ++tk) { at[tq][tk] = __expf(at[tq][tk] - mx); sum += at[tq][tk]; }
            float inv = 1.f / sum;
            for (int tk = 0; tk < 4; ++tk) at[tq][tk] *= inv;
        }
        for (int tq = 0; tq < 4; ++tq)
            for (int d = 0; d < 32; ++d) {
                float acc = 0.f;
                for (int tk = 0; tk < 4; ++tk)
                    acc += at[tq][tk] * s_v[(g * 4 + tk) * DM + hh * 32 + d];
                s_ao[(g * 4 + tq) * DM + hh * 32 + d] = acc;
            }
    }
    __syncthreads();

    // ao2 = ao @ wo + bo + tokens  (into s_q)
    gemm_16x128(s_ao, wo, bo, s_tok, s_q, lane, wvid);
    __syncthreads();

    // LN1 per row -> s_k
    if (tid < 16) {
        float mu = 0.f;
        for (int d = 0; d < DM; ++d) mu += s_q[tid * DM + d];
        mu /= DM;
        float var = 0.f;
        for (int d = 0; d < DM; ++d) { float t = s_q[tid * DM + d] - mu; var += t * t; }
        var /= DM;
        float rstd = rsqrtf(var + 1e-5f);
        for (int d = 0; d < DM; ++d)
            s_k[tid * DM + d] = (s_q[tid * DM + d] - mu) * rstd * ln1_g[d] + ln1_b[d];
    }
    __syncthreads();

    // gate hidden: gelu(g @ gw1 + gb1)  [16][64] -> s_v
    for (int u = tid; u < 16 * 64; u += 128) {
        int r = u >> 6, cc = u & 63;
        float acc = gb1[cc];
        for (int d = 0; d < DM; ++d) acc += s_k[r * DM + d] * gw1[d * 64 + cc];
        s_v[u] = gelu_exact(acc);
    }
    __syncthreads();
    if (tid < 16) {
        float acc = gb2[0];
        for (int c = 0; c < 64; ++c) acc += s_v[tid * 64 + c] * gw2[c];
        s_gate[tid] = acc;
    }
    __syncthreads();
    if (tid < 4) {
        float mx = -3.4e38f;
        for (int t = 0; t < 4; ++t) mx = fmaxf(mx, s_gate[tid * 4 + t]);
        float sum = 0.f, e[4];
        for (int t = 0; t < 4; ++t) { e[t] = __expf(s_gate[tid * 4 + t] - mx); sum += e[t]; }
        for (int t = 0; t < 4; ++t) s_wts[tid * 4 + t] = e[t] / sum;
    }
    __syncthreads();
    // fused = sum_t ao2_t * wts_t
    for (int u = tid; u < 4 * DM; u += 128) {
        int g = u >> 7, d = u & 127;
        float acc = 0.f;
        for (int t = 0; t < 4; ++t) acc += s_q[(g * 4 + t) * DM + d] * s_wts[g * 4 + t];
        s_fused[u] = acc;
    }
    __syncthreads();
    // LN2 per graph
    if (tid < 4) {
        float mu = 0.f;
        for (int d = 0; d < DM; ++d) mu += s_fused[tid * DM + d];
        mu /= DM;
        float var = 0.f;
        for (int d = 0; d < DM; ++d) { float t = s_fused[tid * DM + d] - mu; var += t * t; }
        var /= DM;
        float rstd = rsqrtf(var + 1e-5f);
        for (int d = 0; d < DM; ++d)
            s_fln[tid * DM + d] = (s_fused[tid * DM + d] - mu) * rstd * ln2_g[d] + ln2_b[d];
    }
    __syncthreads();
    // time embedding hidden
    for (int u = tid; u < 4 * 64; u += 128) {
        int g = u >> 6, c = u & 63;
        float ts = srt[mbase + g];
        s_hid[u] = gelu_exact(ts * tw1[c] + tb1[c]);
    }
    __syncthreads();
    // out = LN2(fused) + hid @ tw2 + tb2
    for (int u = tid; u < 4 * DM; u += 128) {
        int g = u >> 7, d = u & 127;
        float acc = tb2[d] + s_fln[g * DM + d];
        for (int c = 0; c < 64; ++c) acc += s_hid[g * 64 + c] * tw2[c * DM + d];
        out[(size_t)(mbase + g) * DM + d] = acc;
    }
}

extern "C" void kernel_launch(void* const* d_in, const int* in_sizes, int n_in,
                              void* d_out, int out_size, void* d_ws, size_t ws_size,
                              hipStream_t stream) {
    const float* bn  = (const float*)d_in[0];
    // d_in[1] = valid_patch_counts (unused by reference forward)
    const float* srt = (const float*)d_in[2];
    const int M = in_sizes[2];  // B*P = 2560

    // params, setup_inputs() insertion order
    const float* gW1[3]; const float* ga1[3]; const float* gW2[3]; const float* ga2[3];
    int pi = 3;
    for (int f = 0; f < 3; ++f) {
        gW1[f] = (const float*)d_in[pi++]; ga1[f] = (const float*)d_in[pi++];
        gW2[f] = (const float*)d_in[pi++]; ga2[f] = (const float*)d_in[pi++];
    }
    const float* cW1 = (const float*)d_in[15]; const float* cb1 = (const float*)d_in[16];
    const float* cW2 = (const float*)d_in[17]; const float* cb2 = (const float*)d_in[18];
    const float* proj_w = (const float*)d_in[19]; const float* proj_b = (const float*)d_in[20];
    const float* wq = (const float*)d_in[21]; const float* bq = (const float*)d_in[22];
    const float* wk = (const float*)d_in[23]; const float* bk = (const float*)d_in[24];
    const float* wv = (const float*)d_in[25]; const float* bv = (const float*)d_in[26];
    const float* wo = (const float*)d_in[27]; const float* bo = (const float*)d_in[28];
    const float* ln1_g = (const float*)d_in[29]; const float* ln1_b = (const float*)d_in[30];
    const float* gw1 = (const float*)d_in[31]; const float* gb1 = (const float*)d_in[32];
    const float* gw2 = (const float*)d_in[33]; const float* gb2 = (const float*)d_in[34];
    const float* ln2_g = (const float*)d_in[35]; const float* ln2_b = (const float*)d_in[36];
    const float* tw1 = (const float*)d_in[37]; const float* tb1 = (const float*)d_in[38];
    const float* tw2 = (const float*)d_in[39]; const float* tb2 = (const float*)d_in[40];

    float* out32 = (float*)d_ws;   // [M,4,32]
    float* out   = (float*)d_out;  // [M,128]

    for (int f = 0; f < 3; ++f)
        gat_kernel<<<M, 256, 0, stream>>>(bn, gW1[f], ga1[f], gW2[f], ga2[f], out32, f);
    gcn_kernel<<<M, 256, 0, stream>>>(bn, cW1, cb1, cW2, cb2, out32);
    fuse_kernel<<<M / 4, 128, 0, stream>>>(out32, proj_w, proj_b, wq, bq, wk, bk, wv, bv,
                                           wo, bo, ln1_g, ln1_b, gw1, gb1, gw2, gb2,
                                           ln2_g, ln2_b, tw1, tb1, tw2, tb2, srt, out);
}
